// PositionAwareMeanAttention_72267119723367
// MI455X (gfx1250) — matrix operand order repaired and checked by reading, not compiled, not run on hardware
//
#include <hip/hip_runtime.h>

typedef __attribute__((ext_vector_type(16))) _Float16 v16h;
typedef __attribute__((ext_vector_type(8)))  float    v8f;

#define T_DIM 200
#define B_DIM 2048
#define H_DIM 128
#define BH    (B_DIM * H_DIM)

// ---------------------------------------------------------------------------
// Pass 1: session_mean[b,h] = sum_t x[t,b,h] * mask[t,b]
// ---------------------------------------------------------------------------
__global__ void __launch_bounds__(256)
mean_kernel(const float* __restrict__ x, const float* __restrict__ mask,
            float* __restrict__ mean) {
    int idx = blockIdx.x * blockDim.x + threadIdx.x;      // over B*H
    if (idx >= BH) return;
    int b = idx >> 7;
    float s = 0.f;
    for (int t = 0; t < T_DIM; ++t) {
        if (t + 8 < T_DIM)
            __builtin_prefetch(x + (long)(t + 8) * BH + idx, 0, 0);
        s += x[(long)t * BH + idx] * mask[t * B_DIM + b];
    }
    mean[idx] = s;
}

// ---------------------------------------------------------------------------
// Shared helper: convert a 128x128 fp32 weight matrix into f16 WMMA B-fragment
// order in LDS.  Fragment (kt,nt), lane L, elem i  ->  B[k][n] with
//   k = kt*32 + (i<8 ? 8*(L>>4)+i : 16+8*(L>>4)+(i-8)),  n = nt*16 + (L&15)
// so each lane's 16 halves are 32 contiguous bytes.
// ---------------------------------------------------------------------------
__device__ inline void fill_wfrag(const float* __restrict__ w, _Float16* lds,
                                  int tid, int nthr) {
    for (int idx = tid; idx < 4 * 8 * 32 * 16; idx += nthr) {
        int i  = idx & 15;
        int L  = (idx >> 4) & 31;
        int nt = (idx >> 9) & 7;
        int kt = idx >> 12;
        int k  = kt * 32 + ((i < 8) ? ((L >> 4) * 8 + i) : (16 + (L >> 4) * 8 + (i - 8)));
        int n  = nt * 16 + (L & 15);
        lds[idx] = (_Float16)w[k * H_DIM + n];
    }
}

// ---------------------------------------------------------------------------
// Pass 2: query = session_mean @ w_q   (2048x128 @ 128x128, fp32 out)
// One wave per 16-row tile; 32 x v_wmma_f32_16x16x32_f16 per tile.
// ---------------------------------------------------------------------------
__global__ void __launch_bounds__(128)
query_gemm_kernel(const float* __restrict__ mean, const float* __restrict__ wq,
                  float* __restrict__ query) {
    __shared__ _Float16 wf[4 * 8 * 32 * 16];              // 32 KB
    fill_wfrag(wq, wf, threadIdx.x, blockDim.x);
    __syncthreads();

    int wave = threadIdx.x >> 5;
    int lane = threadIdx.x & 31;
    int tile = blockIdx.x * (blockDim.x >> 5) + wave;
    if (tile * 16 >= B_DIM) return;
    int b0 = tile * 16;
    int m  = lane & 15;
    int kofs = (lane < 16) ? 0 : 8;
    const float* arow = mean + (long)(b0 + m) * H_DIM;

    v16h af[4];
#pragma unroll
    for (int kt = 0; kt < 4; ++kt) {
        const float* p = arow + kt * 32 + kofs;
#pragma unroll
        for (int i = 0; i < 8; ++i) {
            af[kt][i]     = (_Float16)p[i];
            af[kt][i + 8] = (_Float16)p[16 + i];
        }
    }

    v8f acc[8];
#pragma unroll
    for (int nt = 0; nt < 8; ++nt)
#pragma unroll
        for (int r = 0; r < 8; ++r) acc[nt][r] = 0.f;

#pragma unroll
    for (int nt = 0; nt < 8; ++nt) {
#pragma unroll
        for (int kt = 0; kt < 4; ++kt) {
            const v16h* bp = (const v16h*)&wf[((kt * 8 + nt) * 32 + lane) * 16];
            acc[nt] = __builtin_amdgcn_wmma_f32_16x16x32_f16(
                false, af[kt], false, *bp, (short)0, acc[nt], false, false);
        }
    }

    int rAdd = (lane < 16) ? 0 : 8;
    int dlo  = lane & 15;
#pragma unroll
    for (int nt = 0; nt < 8; ++nt) {
        int d = nt * 16 + dlo;
#pragma unroll
        for (int r = 0; r < 8; ++r)
            query[(long)(b0 + rAdd + r) * H_DIM + d] = acc[nt][r];
    }
}

// ---------------------------------------------------------------------------
// Pass 3 (dominant): for each row g = t*B + b (16-row tiles):
//   key_row = (x[g,:]*mask[t,b]) @ w_k          (WMMA, K=128, N=128)
//   sim[t,b] = sum_d sigmoid(key + query[b,d]) * w_p[t,d]
// key is never written to memory.
// ---------------------------------------------------------------------------
__global__ void __launch_bounds__(128)
score_gemm_kernel(const float* __restrict__ x, const float* __restrict__ mask,
                  const float* __restrict__ wk, const float* __restrict__ query,
                  const float* __restrict__ wp, float* __restrict__ sim) {
    __shared__ _Float16 wf[4 * 8 * 32 * 16];              // 32 KB
    fill_wfrag(wk, wf, threadIdx.x, blockDim.x);
    __syncthreads();

    int wave = threadIdx.x >> 5;
    int lane = threadIdx.x & 31;
    int tile = blockIdx.x * (blockDim.x >> 5) + wave;
    long rowBase = (long)tile * 16;
    if (rowBase >= (long)T_DIM * B_DIM) return;
    int t  = (int)(rowBase / B_DIM);                      // tile never crosses t
    int b0 = (int)(rowBase % B_DIM);

    int m    = lane & 15;                                 // A row this lane feeds
    int kofs = (lane < 16) ? 0 : 8;
    float mval = mask[t * B_DIM + b0 + m];
    const float* xrow = x + (rowBase + m) * H_DIM;

    // A fragments: masked x row, fp32 -> f16
    v16h af[4];
#pragma unroll
    for (int kt = 0; kt < 4; ++kt) {
        const float* p = xrow + kt * 32 + kofs;
#pragma unroll
        for (int i = 0; i < 8; ++i) {
            af[kt][i]     = (_Float16)(p[i] * mval);
            af[kt][i + 8] = (_Float16)(p[16 + i] * mval);
        }
    }

    v8f acc[8];
#pragma unroll
    for (int nt = 0; nt < 8; ++nt)
#pragma unroll
        for (int r = 0; r < 8; ++r) acc[nt][r] = 0.f;

#pragma unroll
    for (int nt = 0; nt < 8; ++nt) {
#pragma unroll
        for (int kt = 0; kt < 4; ++kt) {
            const v16h* bp = (const v16h*)&wf[((kt * 8 + nt) * 32 + lane) * 16];
            acc[nt] = __builtin_amdgcn_wmma_f32_16x16x32_f16(
                false, af[kt], false, *bp, (short)0, acc[nt], false, false);
        }
    }

    // Fused epilogue: sigmoid(key + query) dot w_p[t], row-reduce over d.
    // C/D layout: VGPR r -> (lanes 0-15: M=r, d=nt*16+lane),
    //                       (lanes 16-31: M=8+r, d=nt*16+lane-16)
    int rAdd = (lane < 16) ? 0 : 8;
    int dlo  = lane & 15;
    float psum[8];
#pragma unroll
    for (int r = 0; r < 8; ++r) psum[r] = 0.f;
#pragma unroll
    for (int nt = 0; nt < 8; ++nt) {
        int d = nt * 16 + dlo;
        float wpv = wp[t * H_DIM + d];
#pragma unroll
        for (int r = 0; r < 8; ++r) {
            float v = acc[nt][r] + query[(long)(b0 + rAdd + r) * H_DIM + d];
            float sgm = 1.f / (1.f + __expf(-v));
            psum[r] += sgm * wpv;
        }
    }
#pragma unroll
    for (int r = 0; r < 8; ++r) {
        float v = psum[r];
        v += __shfl_xor(v, 1, 32);                        // reduces within each
        v += __shfl_xor(v, 2, 32);                        // 16-lane half
        v += __shfl_xor(v, 4, 32);
        v += __shfl_xor(v, 8, 32);
        psum[r] = v;
    }
    if (dlo == 0) {                                       // lane 0 and lane 16
#pragma unroll
        for (int r = 0; r < 8; ++r)
            sim[t * B_DIM + b0 + rAdd + r] = psum[r];
    }
}

// ---------------------------------------------------------------------------
// Pass 4: alpha[t,b] = exp(scores - max)*mask / sum(exp(scores - max)*mask)
// (the softmax denominator cancels against the renormalization by p)
// ---------------------------------------------------------------------------
__global__ void __launch_bounds__(256)
alpha_kernel(const float* __restrict__ sim, const float* __restrict__ mask,
             float* __restrict__ alpha) {
    int b = blockIdx.x * blockDim.x + threadIdx.x;
    if (b >= B_DIM) return;
    float mx = -1e30f;
    for (int t = 0; t < T_DIM; ++t) {
        float s = sim[t * B_DIM + b] * mask[t * B_DIM + b];
        mx = fmaxf(mx, s);
    }
    float denom = 0.f;
    for (int t = 0; t < T_DIM; ++t) {
        float mv = mask[t * B_DIM + b];
        float e  = __expf(sim[t * B_DIM + b] * mv - mx) * mv;
        alpha[t * B_DIM + b] = e;
        denom += e;
    }
    float inv = 1.f / denom;
    for (int t = 0; t < T_DIM; ++t) alpha[t * B_DIM + b] *= inv;
}

// ---------------------------------------------------------------------------
// Pass 5: paa_h[b,h] = sum_t x[t,b,h] * alpha[t,b]   (alpha already has mask)
// ---------------------------------------------------------------------------
__global__ void __launch_bounds__(256)
out_kernel(const float* __restrict__ x, const float* __restrict__ alpha,
           float* __restrict__ out) {
    int idx = blockIdx.x * blockDim.x + threadIdx.x;      // over B*H
    if (idx >= BH) return;
    int b = idx >> 7;
    float s = 0.f;
    for (int t = 0; t < T_DIM; ++t) {
        if (t + 8 < T_DIM)
            __builtin_prefetch(x + (long)(t + 8) * BH + idx, 0, 0);
        s += x[(long)t * BH + idx] * alpha[t * B_DIM + b];
    }
    out[idx] = s;
}

// ---------------------------------------------------------------------------
extern "C" void kernel_launch(void* const* d_in, const int* in_sizes, int n_in,
                              void* d_out, int out_size, void* d_ws, size_t ws_size,
                              hipStream_t stream) {
    const float* x    = (const float*)d_in[0];   // [T,B,H]
    const float* mask = (const float*)d_in[1];   // [T,B]
    const float* wp   = (const float*)d_in[2];   // [T,H]
    const float* wk   = (const float*)d_in[3];   // [H,H]
    const float* wq   = (const float*)d_in[4];   // [H,H]
    float* out = (float*)d_out;                  // [B,H]

    float* mean  = (float*)d_ws;                 // B*H
    float* query = mean  + BH;                   // B*H
    float* sim   = query + BH;                   // T*B
    float* alpha = sim   + (size_t)T_DIM * B_DIM;// T*B

    mean_kernel<<<(BH + 255) / 256, 256, 0, stream>>>(x, mask, mean);

    // 2048/16 = 128 tiles, 4 waves (tiles) per 128-thread block
    query_gemm_kernel<<<(B_DIM / 16) / 4, 128, 0, stream>>>(mean, wq, query);

    // 200*2048/16 = 25600 tiles, 4 per block -> 6400 blocks
    score_gemm_kernel<<<(T_DIM * B_DIM / 16) / 4, 128, 0, stream>>>(
        x, mask, wk, query, wp, sim);

    alpha_kernel<<<(B_DIM + 255) / 256, 256, 0, stream>>>(sim, mask, alpha);

    out_kernel<<<(BH + 255) / 256, 256, 0, stream>>>(x, alpha, out);
}
